// EquivariantBlock_77395310674476
// MI455X (gfx1250) — matrix operand order repaired and checked by reading, not compiled, not run on hardware
//
#include <hip/hip_runtime.h>
#include <hip/hip_bf16.h>

// ---------------- problem constants ----------------
#define N_NODES 50000
#define E_EDGES 800000
#define H_DIM   128
#define FE_DIM  16
#define MF_DIM  273          // 2H + FE + 1
#define KS_F    9            // K-steps for 288-wide GEMM (K padded 273 -> 288)
#define KS_H    4            // K-steps for 128-wide GEMM
#define KS_2H   8            // K-steps for 256-wide GEMM
#define NT      8            // N-tiles (128 / 16)

#define FSTRIDE 296          // LDS row stride (elems) for f tile (288 + 8 pad)
#define MSTRIDE 136          // LDS row stride for 128-wide intermediates
#define ASTRIDE 264          // LDS row stride for node concat tile (256 + 8)

typedef __attribute__((ext_vector_type(16))) __bf16 v16bf;
typedef __attribute__((ext_vector_type(8)))  float  v8f;

union Frag { v16bf v; unsigned int u[8]; };

// float -> bf16 (round to nearest even)
__device__ __forceinline__ unsigned short f2bf(float x) {
    union { float f; unsigned int u; } c; c.f = x;
    unsigned int r = c.u + 0x7FFFu + ((c.u >> 16) & 1u);
    return (unsigned short)(r >> 16);
}
__device__ __forceinline__ float bf2f(unsigned int bits16) {
    union { unsigned int u; float f; } c; c.u = bits16 << 16;
    return c.f;
}
__device__ __forceinline__ float sigmoidf_fast(float x) {
    return 1.0f / (1.0f + __expf(-x));
}

// A fragment: 16x32 bf16, row-major in LDS with stride strideElems.
// ISA 7.12.2: lane L holds row m=L%16; half h=L/16; VGPR v holds K pair
// k0 = 16*(v>>2) + 2*(v&3) + 8*h  (+ kbase), packed (k0 lo16, k0+1 hi16).
__device__ __forceinline__ v16bf load_a_frag(const unsigned short* lds,
                                             int strideElems, int kbase, int lane) {
    const int m = lane & 15, h = lane >> 4;
    Frag f;
#pragma unroll
    for (int v = 0; v < 8; ++v) {
        const int k0 = kbase + ((v >> 2) << 4) + ((v & 3) << 1) + (h << 3);
        f.u[v] = *(const unsigned int*)(lds + m * strideElems + k0);
    }
    return f.v;
}

template <int KSTEPS>
__device__ __forceinline__ void load_a_all(const unsigned short* lds, int strideElems,
                                           int lane, v16bf* aF) {
#pragma unroll
    for (int ks = 0; ks < KSTEPS; ++ks) aF[ks] = load_a_frag(lds, strideElems, ks * 32, lane);
}

// B fragment: pre-packed 1024B blob; lane L reads 32 contiguous bytes (2 x b128).
__device__ __forceinline__ v16bf load_b_frag(const unsigned int* packed, int lane) {
    const unsigned int* p = packed + lane * 8;
    Frag f;
#pragma unroll
    for (int v = 0; v < 8; ++v) f.u[v] = p[v];
    return f.v;
}

// C[16x16] f32 fragment layout: lane -> n = lane%16 ; VGPR v -> m = v + 8*(lane/16)

// Two independent accumulation chains (N-tile pair) over one shared A stream.
// B fragments explicitly double-buffered so next-K loads overlap current wmmas.
template <int KSTEPS>
__device__ __forceinline__ void gemm_pair(const v16bf* aF,
                                          const unsigned int* __restrict__ bp0,
                                          const unsigned int* __restrict__ bp1,
                                          float bias0, float bias1,
                                          int lane, v8f& acc0, v8f& acc1) {
#pragma unroll
    for (int v = 0; v < 8; ++v) { acc0[v] = bias0; acc1[v] = bias1; }
    v16bf b0 = load_b_frag(bp0, lane);
    v16bf b1 = load_b_frag(bp1, lane);
#pragma unroll
    for (int ks = 0; ks < KSTEPS; ++ks) {
        v16bf nb0 = b0, nb1 = b1;
        if (ks + 1 < KSTEPS) {
            nb0 = load_b_frag(bp0 + (ks + 1) * 256, lane);
            nb1 = load_b_frag(bp1 + (ks + 1) * 256, lane);
        }
        acc0 = __builtin_amdgcn_wmma_f32_16x16x32_bf16(false, aF[ks], false, b0,
                                                       (short)0, acc0, false, false);
        acc1 = __builtin_amdgcn_wmma_f32_16x16x32_bf16(false, aF[ks], false, b1,
                                                       (short)0, acc1, false, false);
        b0 = nb0; b1 = nb1;
    }
}

// silu(acc) -> bf16 column block nt of a [16 x 128] LDS tile (row stride MSTRIDE)
__device__ __forceinline__ void silu_store_tile(const v8f& acc, unsigned short* outLds,
                                                int nt, int lane) {
    const int n = lane & 15, h = lane >> 4;
#pragma unroll
    for (int v = 0; v < 8; ++v) {
        float x = acc[v];
        x = x * sigmoidf_fast(x);
        outLds[(v + 8 * h) * MSTRIDE + nt * 16 + n] = f2bf(x);
    }
}

// ---------------- init: zero h_agg, coords_out = coords ----------------
__global__ __launch_bounds__(256) void egnn_init(const float* __restrict__ coords,
                                                 float* __restrict__ hagg,
                                                 float* __restrict__ coordsOut) {
    size_t i = (size_t)blockIdx.x * blockDim.x + threadIdx.x;
    if (i < (size_t)N_NODES * H_DIM) hagg[i] = 0.0f;
    if (i < (size_t)N_NODES * 3)     coordsOut[i] = coords[i];
}

// ---------------- weight pack: row-major f32 [Krows x 128] -> B fragments ----
__global__ __launch_bounds__(256) void egnn_pack(const float* __restrict__ w, int Krows,
                                                 int ktiles, unsigned int* __restrict__ out,
                                                 int total) {
    int t = blockIdx.x * 256 + threadIdx.x;
    if (t >= total) return;
    const int v = t & 7;
    const int lane = (t >> 3) & 31;
    const int frag = t >> 8;
    const int nt = frag / ktiles;
    const int ks = frag % ktiles;
    const int n  = nt * 16 + (lane & 15);
    const int h  = lane >> 4;
    const int k0 = ks * 32 + 16 * (v >> 2) + 2 * (v & 3) + 8 * h;
    const float a0 = (k0 < Krows)     ? w[(size_t)k0 * 128 + n]       : 0.0f;
    const float a1 = (k0 + 1 < Krows) ? w[(size_t)(k0 + 1) * 128 + n] : 0.0f;
    out[t] = (unsigned int)f2bf(a0) | ((unsigned int)f2bf(a1) << 16);
}

// ---------------- edge kernel: 1 wave = 16 edges ----------------
__global__ __launch_bounds__(128) void egnn_edge(
    const float* __restrict__ hN, const float* __restrict__ coords,
    const float* __restrict__ aE, const int* __restrict__ src, const int* __restrict__ dst,
    const float* __restrict__ bc1, const float* __restrict__ bc2,
    const float* __restrict__ wc3,
    const float* __restrict__ be1, const float* __restrict__ be2,
    const float* __restrict__ wa,  const float* __restrict__ ba,
    const unsigned int* __restrict__ wc1p, const unsigned int* __restrict__ wc2p,
    const unsigned int* __restrict__ we1p, const unsigned int* __restrict__ we2p,
    float* __restrict__ hagg, float* __restrict__ coordsOut) {

    __shared__ __align__(16) unsigned short fT[4][16 * FSTRIDE];
    __shared__ __align__(16) unsigned short m1[4][16 * MSTRIDE];
    __shared__ float diffT[4][16 * 3];
    __shared__ float rinvT[4][16];
    __shared__ int   dstT[4][16];

    const int wave = threadIdx.x >> 5;
    const int lane = threadIdx.x & 31;
    const int n    = lane & 15;
    const int hh   = lane >> 4;
    const int tile = blockIdx.x * 4 + wave;     // always < 50000
    const int e0   = tile * 16;

    unsigned short* f  = fT[wave];
    unsigned short* mm = m1[wave];
    float* diffL = diffT[wave];
    float* rinvL = rinvT[wave];
    int*   dstL  = dstT[wave];

    // ---- build f tile (16 x 288 bf16) + geometry side data ----
    const int myE   = e0 + n;
    const int mySrc = src[myE];
    const int myDst = dst[myE];
    if (hh == 0) {
        const float sx = coords[(size_t)mySrc * 3 + 0], sy = coords[(size_t)mySrc * 3 + 1],
                    sz = coords[(size_t)mySrc * 3 + 2];
        const float dx0 = coords[(size_t)myDst * 3 + 0], dy0 = coords[(size_t)myDst * 3 + 1],
                    dz0 = coords[(size_t)myDst * 3 + 2];
        const float dx = sx - dx0, dy = sy - dy0, dz = sz - dz0;
        const float rad = dx * dx + dy * dy + dz * dz;
        diffL[n * 3 + 0] = dx; diffL[n * 3 + 1] = dy; diffL[n * 3 + 2] = dz;
        rinvL[n] = 1.0f / (rad + 1.0f);
        dstL[n]  = myDst;
        f[n * FSTRIDE + 256] = f2bf(rad);
    }
#pragma unroll 1
    for (int r = 0; r < 16; ++r) {
        const int s = __shfl(mySrc, r, 32);
        const int d = __shfl(myDst, r, 32);
        const float* hs = hN + (size_t)s * H_DIM;
        const float* hd = hN + (size_t)d * H_DIM;
        unsigned short* fr = f + r * FSTRIDE;
#pragma unroll
        for (int j = 0; j < 4; ++j) {
            fr[j * 32 + lane]       = f2bf(hs[j * 32 + lane]);
            fr[128 + j * 32 + lane] = f2bf(hd[j * 32 + lane]);
        }
        if (lane < 16) {
            fr[257 + lane] = f2bf(aE[(size_t)(e0 + r) * FE_DIM + lane]);
        } else if (lane - 16 < 15) {
            fr[273 + (lane - 16)] = 0;   // zero-pad K 273..287
        }
    }

    // ================= edge_mlp =================
    // GEMM1: silu(f @ we1 + be1) -> mm (bf16)
    {
        v16bf aF[KS_F];
        load_a_all<KS_F>(f, FSTRIDE, lane, aF);
#pragma unroll
        for (int ntp = 0; ntp < 4; ++ntp) {
            const int nt0 = 2 * ntp, nt1 = nt0 + 1;
            v8f a0, a1;
            gemm_pair<KS_F>(aF, we1p + (size_t)nt0 * KS_F * 256,
                                we1p + (size_t)nt1 * KS_F * 256,
                            be1[nt0 * 16 + n], be1[nt1 * 16 + n], lane, a0, a1);
            silu_store_tile(a0, mm, nt0, lane);
            silu_store_tile(a1, mm, nt1, lane);
        }
    }
    // GEMM2: m = silu(mm @ we2 + be2); attention dot accumulated on the fly,
    // silu'ed message kept packed as bf16 (32 VGPRs) until the gate is known.
    float attAcc[8];
#pragma unroll
    for (int v = 0; v < 8; ++v) attAcc[v] = 0.0f;
    unsigned int mpack[NT][4];
    {
        v16bf aM[KS_H];
        load_a_all<KS_H>(mm, MSTRIDE, lane, aM);
#pragma unroll
        for (int ntp = 0; ntp < 4; ++ntp) {
            const int nt0 = 2 * ntp, nt1 = nt0 + 1;
            v8f a0, a1;
            gemm_pair<KS_H>(aM, we2p + (size_t)nt0 * KS_H * 256,
                                we2p + (size_t)nt1 * KS_H * 256,
                            be2[nt0 * 16 + n], be2[nt1 * 16 + n], lane, a0, a1);
            const float w0 = wa[nt0 * 16 + n];
            const float w1 = wa[nt1 * 16 + n];
#pragma unroll
            for (int v = 0; v < 8; ++v) {
                float x0 = a0[v]; x0 = x0 * sigmoidf_fast(x0);
                float x1 = a1[v]; x1 = x1 * sigmoidf_fast(x1);
                attAcc[v] += x0 * w0 + x1 * w1;
                a0[v] = x0; a1[v] = x1;
            }
#pragma unroll
            for (int p = 0; p < 4; ++p) {
                mpack[nt0][p] = (unsigned int)f2bf(a0[2 * p]) |
                                ((unsigned int)f2bf(a0[2 * p + 1]) << 16);
                mpack[nt1][p] = (unsigned int)f2bf(a1[2 * p]) |
                                ((unsigned int)f2bf(a1[2 * p + 1]) << 16);
            }
        }
    }
#pragma unroll
    for (int mask = 1; mask <= 8; mask <<= 1) {
#pragma unroll
        for (int v = 0; v < 8; ++v) attAcc[v] += __shfl_xor(attAcc[v], mask, 32);
    }
    const float bav = ba[0];
    float gate[8];
#pragma unroll
    for (int v = 0; v < 8; ++v) gate[v] = sigmoidf_fast(attAcc[v] + bav);

    // scatter msg_h -> h_agg[dst]
#pragma unroll
    for (int nt = 0; nt < NT; ++nt) {
#pragma unroll
        for (int p = 0; p < 4; ++p) {
            const unsigned int u = mpack[nt][p];
            const int m0 = 2 * p + 8 * hh;
            const float v0 = gate[2 * p]     * bf2f(u & 0xFFFFu);
            const float v1 = gate[2 * p + 1] * bf2f(u >> 16);
            __hip_atomic_fetch_add(&hagg[(size_t)dstL[m0] * H_DIM + nt * 16 + n], v0,
                                   __ATOMIC_RELAXED, __HIP_MEMORY_SCOPE_AGENT);
            __hip_atomic_fetch_add(&hagg[(size_t)dstL[m0 + 1] * H_DIM + nt * 16 + n], v1,
                                   __ATOMIC_RELAXED, __HIP_MEMORY_SCOPE_AGENT);
        }
    }

    // ================= coord_mlp =================
    {
        v16bf aF[KS_F];
        load_a_all<KS_F>(f, FSTRIDE, lane, aF);
#pragma unroll
        for (int ntp = 0; ntp < 4; ++ntp) {
            const int nt0 = 2 * ntp, nt1 = nt0 + 1;
            v8f a0, a1;
            gemm_pair<KS_F>(aF, wc1p + (size_t)nt0 * KS_F * 256,
                                wc1p + (size_t)nt1 * KS_F * 256,
                            bc1[nt0 * 16 + n], bc1[nt1 * 16 + n], lane, a0, a1);
            silu_store_tile(a0, mm, nt0, lane);
            silu_store_tile(a1, mm, nt1, lane);
        }
    }
    float sAcc[8];
#pragma unroll
    for (int v = 0; v < 8; ++v) sAcc[v] = 0.0f;
    {
        v16bf aM[KS_H];
        load_a_all<KS_H>(mm, MSTRIDE, lane, aM);
#pragma unroll
        for (int ntp = 0; ntp < 4; ++ntp) {
            const int nt0 = 2 * ntp, nt1 = nt0 + 1;
            v8f a0, a1;
            gemm_pair<KS_H>(aM, wc2p + (size_t)nt0 * KS_H * 256,
                                wc2p + (size_t)nt1 * KS_H * 256,
                            bc2[nt0 * 16 + n], bc2[nt1 * 16 + n], lane, a0, a1);
            const float w0 = wc3[nt0 * 16 + n];
            const float w1 = wc3[nt1 * 16 + n];
#pragma unroll
            for (int v = 0; v < 8; ++v) {
                float x0 = a0[v]; x0 = x0 * sigmoidf_fast(x0);
                float x1 = a1[v]; x1 = x1 * sigmoidf_fast(x1);
                sAcc[v] += x0 * w0 + x1 * w1;
            }
        }
    }
#pragma unroll
    for (int mask = 1; mask <= 8; mask <<= 1) {
#pragma unroll
        for (int v = 0; v < 8; ++v) sAcc[v] += __shfl_xor(sAcc[v], mask, 32);
    }
    if (n < 3) {
#pragma unroll
        for (int v = 0; v < 8; ++v) {
            const int m = v + 8 * hh;
            const float val = sAcc[v] * diffL[m * 3 + n] * rinvL[m];
            __hip_atomic_fetch_add(&coordsOut[(size_t)dstL[m] * 3 + n], val,
                                   __ATOMIC_RELAXED, __HIP_MEMORY_SCOPE_AGENT);
        }
    }
}

// ---------------- node kernel: 1 wave = 16 nodes ----------------
__global__ __launch_bounds__(128) void egnn_node(
    const float* __restrict__ hN, const float* __restrict__ hagg,
    const float* __restrict__ bn1, const float* __restrict__ bn2,
    const unsigned int* __restrict__ wn1p, const unsigned int* __restrict__ wn2p,
    float* __restrict__ hOut) {

    __shared__ __align__(16) unsigned short aT[4][16 * ASTRIDE];
    __shared__ __align__(16) unsigned short m1[4][16 * MSTRIDE];

    const int wave = threadIdx.x >> 5;
    const int lane = threadIdx.x & 31;
    const int n    = lane & 15;
    const int hh   = lane >> 4;
    const int tile = blockIdx.x * 4 + wave;
    if (tile >= (N_NODES / 16)) return;        // wave-uniform guard
    const int n0 = tile * 16;

    unsigned short* at = aT[wave];
    unsigned short* mm = m1[wave];

    // build concat[h, h_agg] tile (16 x 256 bf16)
#pragma unroll 1
    for (int r = 0; r < 16; ++r) {
        const float* hr = hN   + (size_t)(n0 + r) * H_DIM;
        const float* gr = hagg + (size_t)(n0 + r) * H_DIM;
        unsigned short* ar = at + r * ASTRIDE;
#pragma unroll
        for (int j = 0; j < 4; ++j) {
            ar[j * 32 + lane]       = f2bf(hr[j * 32 + lane]);
            ar[128 + j * 32 + lane] = f2bf(gr[j * 32 + lane]);
        }
    }

    // GEMM1: silu(concat @ wn1 + bn1) -> mm
    {
        v16bf aF[KS_2H];
        load_a_all<KS_2H>(at, ASTRIDE, lane, aF);
#pragma unroll
        for (int ntp = 0; ntp < 4; ++ntp) {
            const int nt0 = 2 * ntp, nt1 = nt0 + 1;
            v8f a0, a1;
            gemm_pair<KS_2H>(aF, wn1p + (size_t)nt0 * KS_2H * 256,
                                 wn1p + (size_t)nt1 * KS_2H * 256,
                             bn1[nt0 * 16 + n], bn1[nt1 * 16 + n], lane, a0, a1);
            silu_store_tile(a0, mm, nt0, lane);
            silu_store_tile(a1, mm, nt1, lane);
        }
    }
    // GEMM2: h_out = h + mm @ wn2 + bn2 (written straight to global per tile)
    {
        v16bf aM[KS_H];
        load_a_all<KS_H>(mm, MSTRIDE, lane, aM);
#pragma unroll
        for (int ntp = 0; ntp < 4; ++ntp) {
            const int nt0 = 2 * ntp, nt1 = nt0 + 1;
            v8f a0, a1;
            gemm_pair<KS_H>(aM, wn2p + (size_t)nt0 * KS_H * 256,
                                wn2p + (size_t)nt1 * KS_H * 256,
                            bn2[nt0 * 16 + n], bn2[nt1 * 16 + n], lane, a0, a1);
#pragma unroll
            for (int v = 0; v < 8; ++v) {
                const int m = v + 8 * hh;
                const size_t i0 = (size_t)(n0 + m) * H_DIM + nt0 * 16 + n;
                const size_t i1 = (size_t)(n0 + m) * H_DIM + nt1 * 16 + n;
                hOut[i0] = hN[i0] + a0[v];
                hOut[i1] = hN[i1] + a1[v];
            }
        }
    }
}

// ---------------- host glue ----------------
extern "C" void kernel_launch(void* const* d_in, const int* in_sizes, int n_in,
                              void* d_out, int out_size, void* d_ws, size_t ws_size,
                              hipStream_t stream) {
    const float* h      = (const float*)d_in[0];
    const float* coords = (const float*)d_in[1];
    const float* a      = (const float*)d_in[2];
    const int*   src    = (const int*)d_in[3];
    const int*   dst    = (const int*)d_in[4];
    const float* wc1 = (const float*)d_in[5];
    const float* bc1 = (const float*)d_in[6];
    const float* wc2 = (const float*)d_in[7];
    const float* bc2 = (const float*)d_in[8];
    const float* wc3 = (const float*)d_in[9];
    const float* we1 = (const float*)d_in[10];
    const float* be1 = (const float*)d_in[11];
    const float* we2 = (const float*)d_in[12];
    const float* be2 = (const float*)d_in[13];
    const float* wa  = (const float*)d_in[14];
    const float* ba  = (const float*)d_in[15];
    const float* wn1 = (const float*)d_in[16];
    const float* bn1 = (const float*)d_in[17];
    const float* wn2 = (const float*)d_in[18];
    const float* bn2 = (const float*)d_in[19];

    float* hOut      = (float*)d_out;
    float* coordsOut = hOut + (size_t)N_NODES * H_DIM;

    // workspace layout
    char* ws = (char*)d_ws;
    float* hagg = (float*)ws;
    size_t off = (size_t)N_NODES * H_DIM * sizeof(float);
    const int DW_F  = NT * KS_F  * 256;  // 18432 dwords for 288-wide weights
    const int DW_H  = NT * KS_H  * 256;  //  8192 dwords for 128-wide weights
    const int DW_2H = NT * KS_2H * 256;  // 16384 dwords for 256-wide weights
    unsigned int* we1p = (unsigned int*)(ws + off); off += (size_t)DW_F  * 4;
    unsigned int* wc1p = (unsigned int*)(ws + off); off += (size_t)DW_F  * 4;
    unsigned int* we2p = (unsigned int*)(ws + off); off += (size_t)DW_H  * 4;
    unsigned int* wc2p = (unsigned int*)(ws + off); off += (size_t)DW_H  * 4;
    unsigned int* wn1p = (unsigned int*)(ws + off); off += (size_t)DW_2H * 4;
    unsigned int* wn2p = (unsigned int*)(ws + off); off += (size_t)DW_H  * 4;

    // 1) init h_agg = 0, coords_out = coords
    {
        size_t total = (size_t)N_NODES * H_DIM;
        int blocks = (int)((total + 255) / 256);
        egnn_init<<<blocks, 256, 0, stream>>>(coords, hagg, coordsOut);
    }
    // 2) pack weights into WMMA B-fragment order (bf16)
    egnn_pack<<<(DW_F  + 255) / 256, 256, 0, stream>>>(we1, MF_DIM,    KS_F,  we1p, DW_F);
    egnn_pack<<<(DW_F  + 255) / 256, 256, 0, stream>>>(wc1, MF_DIM,    KS_F,  wc1p, DW_F);
    egnn_pack<<<(DW_H  + 255) / 256, 256, 0, stream>>>(we2, H_DIM,     KS_H,  we2p, DW_H);
    egnn_pack<<<(DW_H  + 255) / 256, 256, 0, stream>>>(wc2, H_DIM,     KS_H,  wc2p, DW_H);
    egnn_pack<<<(DW_2H + 255) / 256, 256, 0, stream>>>(wn1, 2 * H_DIM, KS_2H, wn1p, DW_2H);
    egnn_pack<<<(DW_H  + 255) / 256, 256, 0, stream>>>(wn2, H_DIM,     KS_H,  wn2p, DW_H);

    // 3) edge kernel: 50000 tiles of 16 edges, 4 waves/block
    egnn_edge<<<E_EDGES / 16 / 4, 128, 0, stream>>>(
        h, coords, a, src, dst,
        bc1, bc2, wc3, be1, be2, wa, ba,
        wc1p, wc2p, we1p, we2p, hagg, coordsOut);

    // 4) node kernel: 3125 tiles of 16 nodes
    {
        int tiles = N_NODES / 16;
        int blocks = (tiles + 3) / 4;
        egnn_node<<<blocks, 128, 0, stream>>>(h, hagg, bn1, bn2, wn1p, wn2p, hOut);
    }
}
